// EncoderLayer_33019708572120
// MI455X (gfx1250) — matrix operand order repaired and checked by reading, not compiled, not run on hardware
//
#include <hip/hip_runtime.h>
#include <hip/hip_bf16.h>

typedef _Float16 v16h __attribute__((ext_vector_type(16)));
typedef _Float16 v8h  __attribute__((ext_vector_type(8)));
typedef _Float16 v4h  __attribute__((ext_vector_type(4)));
typedef float    v8f  __attribute__((ext_vector_type(8)));
typedef unsigned int v4u __attribute__((ext_vector_type(4)));
typedef int          v4i __attribute__((ext_vector_type(4)));
typedef int          v8i __attribute__((ext_vector_type(8)));

#define D_MODEL 512
#define N_HEADS 8
#define D_FF    2048
#define BATCH   2
#define SEQ     2048
#define DK      64

#if defined(__HIP_DEVICE_COMPILE__) && __has_builtin(__builtin_amdgcn_tensor_load_to_lds)
#define HAVE_TDM 1
#endif

// ---------------------------------------------------------------- helpers

__device__ __forceinline__ v16h cat8(v8h lo, v8h hi) {
  v16h r;
#pragma unroll
  for (int i = 0; i < 8; ++i) { r[i] = lo[i]; r[i + 8] = hi[i]; }
  return r;
}

#ifdef HAVE_TDM
// Issue a TDM 2D tile copy: global (row-major, 2-byte elems) -> LDS.
// D# built per CDNA5 ISA sec 8.3/8.4. tile_w/tile_h in elements, row_stride in elements.
__device__ __forceinline__ void tdm_load_2d(unsigned lds_off, const void* gaddr,
                                            int tile_w, int tile_h, int row_stride) {
  unsigned long long ga = (unsigned long long)(size_t)gaddr;
  v4u g0;
  g0[0] = 1u;                                   // count=1, is_restore=0, gather off
  g0[1] = lds_off;                              // lds_addr (bytes)
  g0[2] = (unsigned)(ga & 0xFFFFFFFFu);         // global_addr[31:0]
  g0[3] = (unsigned)((ga >> 32) & 0x1FFFFFFu)   // global_addr[56:32]
        | (2u << 30);                           // type = 2 ("image")
  v8i g1;
  g1[0] = (1 << 16);                            // wg_mask=0, data_size=1 (2 bytes)
  g1[1] = (tile_w & 0xFFFF) << 16;              // tensor_dim0[15:0]
  g1[2] = ((unsigned)tile_w >> 16) | ((tile_h & 0xFFFF) << 16);   // dim0 hi | tensor_dim1 lo
  g1[3] = ((unsigned)tile_h >> 16) | ((tile_w & 0xFFFF) << 16);   // dim1 hi | tile_dim0
  g1[4] = (tile_h & 0xFFFF);                    // tile_dim1 (tile_dim2 = 0)
  g1[5] = row_stride;                           // tensor_dim0_stride[31:0]
  g1[6] = 0;                                    // stride hi bits / dim1_stride lo
  g1[7] = 0;
  v4i z4 = {0, 0, 0, 0};
#if __clang_major__ >= 23
  v8i z8 = {0, 0, 0, 0, 0, 0, 0, 0};
  __builtin_amdgcn_tensor_load_to_lds(g0, g1, z4, z4, z8, 0);
#else
  __builtin_amdgcn_tensor_load_to_lds(g0, g1, z4, z4, 0);
#endif
}
#endif

// ---------------------------------------------------------------- f32 -> f16 (4 elems/thread)

__global__ void cvt_f32_f16(const float* __restrict__ src, _Float16* __restrict__ dst, int n) {
  int i = (blockIdx.x * blockDim.x + threadIdx.x) * 4;
  if (i < n) {
    float4 f = *(const float4*)(src + i);
    v4h o;
    o[0] = (_Float16)f.x; o[1] = (_Float16)f.y; o[2] = (_Float16)f.z; o[3] = (_Float16)f.w;
    *(v4h*)(dst + i) = o;
  }
}

// ---------------------------------------------------------------- GEMM: C = A @ W^T + bias
// A:[M,K] f16 row-major, W:[N,K] f16 row-major. 128 threads = 4 waves;
// each wave computes a 32x64 tile (2x4 WMMA frags, 8 WMMAs per 32-K step) so each
// A fragment is reused 4x. Block covers 64x128. Compile-time epilogue flags.

template <int RELU, int WRITE_F32, int WRITE_F16>
__global__ void gemm_bias(const _Float16* __restrict__ A, const _Float16* __restrict__ W,
                          const float* __restrict__ bias,
                          float* __restrict__ Cf, _Float16* __restrict__ Ch,
                          int M, int N, int K)
{
  const int wave = threadIdx.x >> 5;
  const int lane = threadIdx.x & 31;
  const int hl   = lane >> 4;        // half-wave id
  const int l16  = lane & 15;
  const int m0 = blockIdx.x * 64 + (wave & 1) * 32;
  const int n0 = blockIdx.y * 128 + (wave >> 1) * 64;

  v8f acc[2][4];
#pragma unroll
  for (int i = 0; i < 2; ++i)
#pragma unroll
    for (int j = 0; j < 4; ++j) acc[i][j] = (v8f)0.0f;

  for (int k0 = 0; k0 < K; k0 += 32) {
    v16h a[2], b[4];
    // A fragment: 16x32, lane row m0+l16, contiguous K run of 16 at k0 + hl*16
#pragma unroll
    for (int mf = 0; mf < 2; ++mf) {
      const _Float16* ap = A + (size_t)(m0 + mf * 16 + l16) * K + k0 + hl * 16;
      a[mf] = cat8(*(const v8h*)ap, *(const v8h*)(ap + 8));
    }
    // B fragment: 32x16, lane col n0+nf*16+l16, contiguous K run of 16 at k0 + hl*16
#pragma unroll
    for (int nf = 0; nf < 4; ++nf) {
      const _Float16* wp = W + (size_t)(n0 + nf * 16 + l16) * K + k0 + hl * 16;
      b[nf] = *(const v16h*)wp;
    }
    // speculative prefetch of next K tile (harmless overshoot on last iteration)
    __builtin_prefetch(A + (size_t)(m0 + l16) * K + k0 + 32 + hl * 8, 0, 3);
    __builtin_prefetch(W + (size_t)(n0 + l16) * K + k0 + 32 + hl * 16, 0, 3);
#pragma unroll
    for (int mf = 0; mf < 2; ++mf)
#pragma unroll
      for (int nf = 0; nf < 4; ++nf)
        acc[mf][nf] = __builtin_amdgcn_wmma_f32_16x16x32_f16(
            false, a[mf], false, b[nf], (short)0, acc[mf][nf], false, false);
  }

  // epilogue: C layout — lane col = n0+nf*16+l16, row = m0+mf*16 + r + hl*8
#pragma unroll
  for (int mf = 0; mf < 2; ++mf) {
#pragma unroll
    for (int nf = 0; nf < 4; ++nf) {
      const int col = n0 + nf * 16 + l16;
      const float bv = bias[col];
#pragma unroll
      for (int r = 0; r < 8; ++r) {
        const int row = m0 + mf * 16 + hl * 8 + r;
        float v = acc[mf][nf][r] + bv;
        if (RELU) v = fmaxf(v, 0.0f);
        if (WRITE_F32) Cf[(size_t)row * N + col] = v;
        if (WRITE_F16) Ch[(size_t)row * N + col] = (_Float16)v;
      }
    }
  }
}

// ---------------------------------------------------------------- flash attention (transposed)
// Q,K,V,Ctx: f16 [B, L, H*DK]. One block per (b*H+h, 64 query rows); 4 waves x 16 rows.
// Computes S^T = K Q^T and O^T = V^T P^T so that softmax rows live *within* a lane:
// per-lane scalar m/l state, single xor-16 shuffle per reduction, no P LDS round trip.
// K tile staged by the Tensor Data Mover; V tile software-pipelined through registers.

__global__ void flash_attn(const _Float16* __restrict__ Q, const _Float16* __restrict__ Km,
                           const _Float16* __restrict__ V, _Float16* __restrict__ Ctx,
                           float scale)
{
  const int L = SEQ, D = D_MODEL;
  const int b = blockIdx.x >> 3, h = blockIdx.x & 7;
  const int wave = threadIdx.x >> 5;
  const int lane = threadIdx.x & 31;
  const int hl = lane >> 4, l16 = lane & 15;
  const int qrow0 = blockIdx.y * 64 + wave * 16;
  const size_t headoff = (size_t)b * L * D + h * DK;

  __shared__ _Float16 ldsK[32 * DK];        // K tile, row-major [s][d], 4 KB
  __shared__ _Float16 ldsVT[DK * 32];       // V^T tile: [d][s], 4 KB

  // Q as B-fragments: lane col q = qrow0+l16, contiguous d-run of 16 (held in regs all kernel)
  v16h bQ[2];
#pragma unroll
  for (int kk = 0; kk < 2; ++kk)
    bQ[kk] = *(const v16h*)(Q + headoff + (size_t)(qrow0 + l16) * D + kk * 32 + hl * 16);

  v8f o[4];
#pragma unroll
  for (int f = 0; f < 4; ++f) o[f] = (v8f)0.0f;
  float mrow = -INFINITY, lrow = 0.0f;

  // software-pipelined V tile: 2 chunks of 8 halves per thread
  const int c0 = threadIdx.x, c1 = threadIdx.x + 128;
  const int s0r0 = c0 >> 3, d00 = (c0 & 7) * 8;
  const int s0r1 = c1 >> 3, d01 = (c1 & 7) * 8;
  v8h vbuf0 = *(const v8h*)(V + headoff + (size_t)s0r0 * D + d00);
  v8h vbuf1 = *(const v8h*)(V + headoff + (size_t)s0r1 * D + d01);

  for (int s0 = 0; s0 < L; s0 += 32) {
    __syncthreads();   // previous iteration's LDS reads complete before overwrite

#ifdef HAVE_TDM
    if (threadIdx.x < 32) {   // one wave issues the TDM copy (TDM ignores EXEC)
      tdm_load_2d((unsigned)(size_t)&ldsK[0], Km + headoff + (size_t)s0 * D,
                  DK /*tile_w*/, 32 /*tile_h*/, D /*row stride*/);
    }
#else
    for (int c = threadIdx.x; c < 32 * 8; c += 128) {
      const int srow = c >> 3;
      const int d0 = (c & 7) * 8;
      *(v8h*)&ldsK[srow * DK + d0] = *(const v8h*)(Km + headoff + (size_t)(s0 + srow) * D + d0);
    }
#endif

    // scatter prefetched V chunks into LDS transposed
#pragma unroll
    for (int i = 0; i < 8; ++i) ldsVT[(d00 + i) * 32 + s0r0] = vbuf0[i];
#pragma unroll
    for (int i = 0; i < 8; ++i) ldsVT[(d01 + i) * 32 + s0r1] = vbuf1[i];

#ifdef HAVE_TDM
    if (threadIdx.x < 32) __builtin_amdgcn_s_wait_tensorcnt(0);
#endif
    __syncthreads();

    // prefetch next V tile into registers (overlaps the math below)
    const int sn = (s0 + 32 < L) ? s0 + 32 : 0;
    vbuf0 = *(const v8h*)(V + headoff + (size_t)(sn + s0r0) * D + d00);
    vbuf1 = *(const v8h*)(V + headoff + (size_t)(sn + s0r1) * D + d01);

    // S^T = K @ Q^T : C-frag sf has rows s = s0 + sf*16 + hl*8 + r, col q = l16
    v8f sc[2];
    sc[0] = (v8f)0.0f; sc[1] = (v8f)0.0f;
#pragma unroll
    for (int sf = 0; sf < 2; ++sf) {
#pragma unroll
      for (int kk = 0; kk < 2; ++kk) {
        const _Float16* kp = &ldsK[(sf * 16 + l16) * DK + kk * 32 + hl * 16];
        v16h aK = cat8(*(const v8h*)kp, *(const v8h*)(kp + 8));
        sc[sf] = __builtin_amdgcn_wmma_f32_16x16x32_f16(
            false, aK, false, bQ[kk], (short)0, sc[sf], false, false);
      }
    }

    // scaled scores; per-lane max over 16 regs + one cross-half shuffle
    float lmax = -INFINITY;
#pragma unroll
    for (int sf = 0; sf < 2; ++sf)
#pragma unroll
      for (int r = 0; r < 8; ++r) {
        sc[sf][r] *= scale;
        lmax = fmaxf(lmax, sc[sf][r]);
      }
    lmax = fmaxf(lmax, __shfl_xor(lmax, 16, 32));
    const float mn = fmaxf(mrow, lmax);
    const float alpha = __expf(mrow - mn);
    mrow = mn;

    float p0[8], p1[8], lsum = 0.0f;
#pragma unroll
    for (int r = 0; r < 8; ++r) {
      p0[r] = __expf(sc[0][r] - mn);
      p1[r] = __expf(sc[1][r] - mn);
      lsum += p0[r] + p1[r];
    }
    lsum += __shfl_xor(lsum, 16, 32);
    lrow = lrow * alpha + lsum;
#pragma unroll
    for (int f = 0; f < 4; ++f)
#pragma unroll
      for (int r = 0; r < 8; ++r) o[f][r] *= alpha;

    // build P^T B-fragment (32 s x 16 q) via cross-half register exchange:
    // lane hl=0 holds s=0..7 (own p0) | s=8..15 (partner p0)
    // lane hl=1 holds s=16..23 (partner p1) | s=24..31 (own p1)
    v16h bP;
#pragma unroll
    for (int r = 0; r < 8; ++r) {
      float x0 = __shfl_xor(p0[r], 16, 32);
      float x1 = __shfl_xor(p1[r], 16, 32);
      bP[r]     = (_Float16)(hl ? x1 : p0[r]);
      bP[r + 8] = (_Float16)(hl ? p1[r] : x0);
    }

    // O^T += V^T @ P^T : A-frag f covers d = f*16..f*16+15
#pragma unroll
    for (int f = 0; f < 4; ++f) {
      const _Float16* vt = &ldsVT[(f * 16 + l16) * 32 + hl * 16];
      v16h aV = cat8(*(const v8h*)vt, *(const v8h*)(vt + 8));
      o[f] = __builtin_amdgcn_wmma_f32_16x16x32_f16(
          false, aV, false, bP, (short)0, o[f], false, false);
    }
  }

  // normalize and store: lane holds query row q = qrow0+l16, d = f*16 + hl*8 + r (contiguous)
  const float inv = 1.0f / lrow;
#pragma unroll
  for (int f = 0; f < 4; ++f) {
    v8h out;
#pragma unroll
    for (int r = 0; r < 8; ++r) out[r] = (_Float16)(o[f][r] * inv);
    *(v8h*)(Ctx + headoff + (size_t)(qrow0 + l16) * D + f * 16 + hl * 8) = out;
  }
}

// ---------------------------------------------------------------- residual + LayerNorm (one 128-thread block per row of 512)

template <int WRITE_F16>
__global__ void add_layernorm(const float* __restrict__ X, const float* __restrict__ Y,
                              const float* __restrict__ g, const float* __restrict__ be,
                              float* __restrict__ Out, _Float16* __restrict__ OutH)
{
  const int D = D_MODEL;
  const int row = blockIdx.x;
  const int t = threadIdx.x;
  const float* x = X + (size_t)row * D;
  const float* y = Y + (size_t)row * D;

  float v[4], s = 0.0f, s2 = 0.0f;
#pragma unroll
  for (int i = 0; i < 4; ++i) {
    const int c = t + i * 128;
    float val = x[c] + y[c];
    v[i] = val; s += val; s2 += val * val;
  }
#pragma unroll
  for (int m = 1; m <= 16; m <<= 1) { s += __shfl_xor(s, m, 32); s2 += __shfl_xor(s2, m, 32); }

  __shared__ float red[8];
  const int wave = t >> 5, lane = t & 31;
  if (lane == 0) { red[wave] = s; red[4 + wave] = s2; }
  __syncthreads();
  float S = 0.0f, S2 = 0.0f;
#pragma unroll
  for (int w = 0; w < 4; ++w) { S += red[w]; S2 += red[4 + w]; }

  const float mean = S * (1.0f / D);
  const float var = S2 * (1.0f / D) - mean * mean;
  const float inv = rsqrtf(var + 1e-5f);
#pragma unroll
  for (int i = 0; i < 4; ++i) {
    const int c = t + i * 128;
    float o = (v[i] - mean) * inv * g[c] + be[c];
    Out[(size_t)row * D + c] = o;
    if (WRITE_F16) OutH[(size_t)row * D + c] = (_Float16)o;
  }
}

// ---------------------------------------------------------------- launch

static inline void cvt(const float* s, _Float16* d, int n, hipStream_t st) {
  cvt_f32_f16<<<(n / 4 + 255) / 256, 256, 0, st>>>(s, d, n);
}

extern "C" void kernel_launch(void* const* d_in, const int* in_sizes, int n_in,
                              void* d_out, int out_size, void* d_ws, size_t ws_size,
                              hipStream_t stream) {
  (void)in_sizes; (void)n_in; (void)out_size; (void)ws_size;
  const float* x   = (const float*)d_in[0];
  const float* Wq  = (const float*)d_in[1];
  const float* bq  = (const float*)d_in[2];
  const float* Wk  = (const float*)d_in[3];
  const float* bk  = (const float*)d_in[4];
  const float* Wv  = (const float*)d_in[5];
  const float* bv  = (const float*)d_in[6];
  const float* Wo  = (const float*)d_in[7];
  const float* bo  = (const float*)d_in[8];
  const float* W1  = (const float*)d_in[9];
  const float* b1  = (const float*)d_in[10];
  const float* W2  = (const float*)d_in[11];
  const float* b2  = (const float*)d_in[12];
  const float* g1  = (const float*)d_in[13];
  const float* be1 = (const float*)d_in[14];
  const float* g2  = (const float*)d_in[15];
  const float* be2 = (const float*)d_in[16];

  const int M = BATCH * SEQ;                 // 4096 rows
  char* ws = (char*)d_ws;
  size_t off = 0;
  auto alloc = [&](size_t bytes) -> void* {
    void* p = ws + off;
    off += (bytes + 255) & ~(size_t)255;
    return p;
  };

  _Float16* xh   = (_Float16*)alloc((size_t)M * D_MODEL * 2);
  _Float16* Wqh  = (_Float16*)alloc((size_t)D_MODEL * D_MODEL * 2);
  _Float16* Wkh  = (_Float16*)alloc((size_t)D_MODEL * D_MODEL * 2);
  _Float16* Wvh  = (_Float16*)alloc((size_t)D_MODEL * D_MODEL * 2);
  _Float16* Woh  = (_Float16*)alloc((size_t)D_MODEL * D_MODEL * 2);
  _Float16* W1h  = (_Float16*)alloc((size_t)D_FF * D_MODEL * 2);
  _Float16* W2h  = (_Float16*)alloc((size_t)D_MODEL * D_FF * 2);
  _Float16* qh   = (_Float16*)alloc((size_t)M * D_MODEL * 2);
  _Float16* kh   = (_Float16*)alloc((size_t)M * D_MODEL * 2);
  _Float16* vh   = (_Float16*)alloc((size_t)M * D_MODEL * 2);
  _Float16* ctxh = (_Float16*)alloc((size_t)M * D_MODEL * 2);
  float*    attn = (float*)alloc((size_t)M * D_MODEL * 4);
  float*    x1   = (float*)alloc((size_t)M * D_MODEL * 4);
  _Float16* x1h  = (_Float16*)alloc((size_t)M * D_MODEL * 2);
  _Float16* yh   = (_Float16*)alloc((size_t)M * D_FF * 2);
  float*    y2   = (float*)alloc((size_t)M * D_MODEL * 4);

  // 1) f16 conversions
  cvt(x,  xh,  M * D_MODEL, stream);
  cvt(Wq, Wqh, D_MODEL * D_MODEL, stream);
  cvt(Wk, Wkh, D_MODEL * D_MODEL, stream);
  cvt(Wv, Wvh, D_MODEL * D_MODEL, stream);
  cvt(Wo, Woh, D_MODEL * D_MODEL, stream);
  cvt(W1, W1h, D_FF * D_MODEL, stream);
  cvt(W2, W2h, D_MODEL * D_FF, stream);

  const dim3 blk(128);
  // 2) Q/K/V projections (f16 out)
  {
    dim3 grd(M / 64, D_MODEL / 128);
    gemm_bias<0, 0, 1><<<grd, blk, 0, stream>>>(xh, Wqh, bq, nullptr, qh, M, D_MODEL, D_MODEL);
    gemm_bias<0, 0, 1><<<grd, blk, 0, stream>>>(xh, Wkh, bk, nullptr, kh, M, D_MODEL, D_MODEL);
    gemm_bias<0, 0, 1><<<grd, blk, 0, stream>>>(xh, Wvh, bv, nullptr, vh, M, D_MODEL, D_MODEL);
  }
  // 3) attention
  {
    dim3 grd(BATCH * N_HEADS, SEQ / 64);
    flash_attn<<<grd, blk, 0, stream>>>(qh, kh, vh, ctxh, 0.125f /* 1/sqrt(64) */);
  }
  // 4) output projection (f32 out)
  {
    dim3 grd(M / 64, D_MODEL / 128);
    gemm_bias<0, 1, 0><<<grd, blk, 0, stream>>>(ctxh, Woh, bo, attn, nullptr, M, D_MODEL, D_MODEL);
  }
  // 5) residual + LN1
  add_layernorm<1><<<M, blk, 0, stream>>>(x, attn, g1, be1, x1, x1h);
  // 6) FFN up + ReLU (f16 out)
  {
    dim3 grd(M / 64, D_FF / 128);
    gemm_bias<1, 0, 1><<<grd, blk, 0, stream>>>(x1h, W1h, b1, nullptr, yh, M, D_FF, D_MODEL);
  }
  // 7) FFN down (f32 out)
  {
    dim3 grd(M / 64, D_MODEL / 128);
    gemm_bias<0, 1, 0><<<grd, blk, 0, stream>>>(yh, W2h, b2, y2, nullptr, M, D_MODEL, D_FF);
  }
  // 8) residual + LN2 -> output (f32)
  add_layernorm<0><<<M, blk, 0, stream>>>(x1, y2, g2, be2, (float*)d_out, nullptr);
}
